// E2ESWIGLULayer_57913339019654
// MI455X (gfx1250) — matrix operand order repaired and checked by reading, not compiled, not run on hardware
//
#include <hip/hip_runtime.h>
#include <hip/hip_bf16.h>

typedef __bf16 bf16;
typedef __attribute__((ext_vector_type(16))) __bf16 bf16x16;
typedef __attribute__((ext_vector_type(8)))  __bf16 bf16x8;
typedef __attribute__((ext_vector_type(8)))  float  f32x8;

#define S_LEN 2048
#define HID   2048
#define NHEAD 16
#define HDIM  128
#define WIN   1024

// ---------------- CDNA5 async copy (global -> LDS, ASYNCcnt-tracked) ----------------
// ldsoff: low 32 bits of the generic shared-memory address == byte offset within the
// workgroup LDS allocation (ISA 10.2: LDS aperture keyed off ADDR[63:32], hardware
// uses the low 32 bits as the LDS address). Deriving it from the real &smem[...] address makes the
// shared array escape, so the "memory" clobber forces the compiler to treat these as
// writes to smem (keeping the ds_load fragment reads alive).
#define ASYNC_CP_B128(ldsoff, gptr)                                              \
  asm volatile("global_load_async_to_lds_b128 %0, %1, off" ::"v"(ldsoff),        \
               "v"(gptr)                                                         \
               : "memory")
#define WAIT_ASYNC_LE8() asm volatile("s_wait_asynccnt 0x8" ::: "memory")
#define WAIT_ASYNC_0()   asm volatile("s_wait_asynccnt 0x0" ::: "memory")

__device__ __forceinline__ unsigned lds_off_u32(const void* p) {
  return (unsigned)(size_t)p;   // low 32 bits of generic shared address = LDS offset
}

// ---------------- WMMA helpers (CDNA5 16x16x32 bf16, f32 accum) ----------------

__device__ __forceinline__ f32x8 wmma_bf16(bf16x16 a, bf16x16 b, f32x8 c) {
  return __builtin_amdgcn_wmma_f32_16x16x32_bf16(false, a, false, b, (short)0, c,
                                                 false, false);
}

// A fragment: 16x32 (MxK) from row-major [*, ld] at (row0, k0).
// lanes 0-15: M=lane, K in {0..7}U{16..23}; lanes 16-31: M=lane-16, K in {8..15}U{24..31}
__device__ __forceinline__ bf16x16 load_a_frag(const bf16* base, int ld, int row0, int k0) {
  int lane = threadIdx.x & 31;
  int lr = lane & 15, half = lane >> 4;
  const bf16* p = base + (size_t)(row0 + lr) * ld + k0 + half * 8;
  bf16x8 c0 = *(const bf16x8*)p;
  bf16x8 c1 = *(const bf16x8*)(p + 16);
  bf16x16 r;
#pragma unroll
  for (int i = 0; i < 8; i++) { r[i] = c0[i]; r[i + 8] = c1[i]; }
  return r;
}

// B fragment: 32x16 (KxN) from [N,K] row-major source at (n0, k0).
// lanes 0-15: N=lane, K in {0..7}U{16..23}; lanes 16-31: N=lane-16, K in {8..15}U{24..31}
__device__ __forceinline__ bf16x16 load_b_frag(const bf16* base, int ld, int n0, int k0) {
  int lane = threadIdx.x & 31;
  int lr = lane & 15, half = lane >> 4;
  const bf16* p = base + (size_t)(n0 + lr) * ld + k0 + half * 8;
  bf16x8 c0 = *(const bf16x8*)p;
  bf16x8 c1 = *(const bf16x8*)(p + 16);
  bf16x16 r;
#pragma unroll
  for (int i = 0; i < 8; i++) { r[i] = c0[i]; r[i + 8] = c1[i]; }
  return r;
}

// ---------------- Kernel 1: fp32 -> bf16 ----------------

__global__ void swa_cvt_bf16(const float* __restrict__ src, bf16* __restrict__ dst, int n) {
  int i = blockIdx.x * 256 + threadIdx.x;
  if (i < n) dst[i] = (bf16)src[i];
}

// ---------------- Kernel 2/5: GEMM  C[M,N] = A[M,K] * W[N,K]^T ----------------
// block = 256 threads = 8 waves; block tile 128x128; wave tile 64x32 (4x2 WMMA tiles).
// K-step 64, double-buffered LDS filled by global_load_async_to_lds_b128.
// LDS tiles padded to pitch 72 elems (144 B) -> conflict-free b128 fragment reads.

__global__ __launch_bounds__(256) void swa_gemm_bf16(const bf16* __restrict__ A,
                                                     const bf16* __restrict__ W,
                                                     float* __restrict__ C,
                                                     int M, int N, int K) {
  __shared__ bf16 smem[2][2][128 * 72];           // [buf][A|B][row*72]
  const unsigned BUFB = 2u * 128u * 72u * 2u;     // 36864 bytes per buffer slot

  int t = threadIdx.x;
  int w = t >> 5;
  int m_blk = blockIdx.y * 128;
  int n_blk = blockIdx.x * 128;

  // per-thread async-copy plan: 8 x 16B chunks (A tile: ids 0..1023, B tile: 1024..2047)
  const bf16* gsrc[8];
  unsigned loff[8];
#pragma unroll
  for (int i = 0; i < 8; i++) {
    int id = i * 256 + t;
    int row = (id & 1023) >> 3, c = id & 7;
    if (id < 1024) {
      gsrc[i] = A + (size_t)(m_blk + row) * K + c * 8;
      loff[i] = lds_off_u32(&smem[0][0][row * 72 + c * 8]);
    } else {
      gsrc[i] = W + (size_t)(n_blk + row) * K + c * 8;
      loff[i] = lds_off_u32(&smem[0][1][row * 72 + c * 8]);
    }
  }

  int m0l = (w >> 2) * 64;
  int n0l = (w & 3) * 32;
  f32x8 acc[4][2] = {};

  // prologue: fill buffer 0 (8 outstanding per wave)
#pragma unroll
  for (int i = 0; i < 8; i++) ASYNC_CP_B128(loff[i], gsrc[i]);

  int cur = 0;
  for (int kt = 0; kt < K; kt += 64) {
    int nk = kt + 64;
    if (nk < K) {
#pragma unroll
      for (int i = 0; i < 8; i++)
        ASYNC_CP_B128(loff[i] + (unsigned)(cur ^ 1) * BUFB, gsrc[i] + nk);
      WAIT_ASYNC_LE8();   // 16 outstanding -> <=8 retires current buffer (in-order)
    } else {
      WAIT_ASYNC_0();
    }
    __syncthreads();
    const bf16* Ab = &smem[cur][0][0];
    const bf16* Bb = &smem[cur][1][0];
#pragma unroll
    for (int ks = 0; ks < 64; ks += 32) {
      bf16x16 a[4], b[2];
#pragma unroll
      for (int i = 0; i < 4; i++) a[i] = load_a_frag(Ab, 72, m0l + 16 * i, ks);
#pragma unroll
      for (int j = 0; j < 2; j++) b[j] = load_b_frag(Bb, 72, n0l + 16 * j, ks);
#pragma unroll
      for (int i = 0; i < 4; i++)
#pragma unroll
        for (int j = 0; j < 2; j++)
          acc[i][j] = wmma_bf16(a[i], b[j], acc[i][j]);
    }
    __syncthreads();    // all reads of the soon-to-be-overwritten buffer complete
    cur ^= 1;
  }

  int lane = threadIdx.x & 31, lr = lane & 15, half = lane >> 4;
  int m0 = m_blk + m0l, n0 = n_blk + n0l;
#pragma unroll
  for (int i = 0; i < 4; i++)
#pragma unroll
    for (int j = 0; j < 2; j++)
#pragma unroll
      for (int v = 0; v < 8; v++) {
        int row = m0 + 16 * i + v + 8 * half;
        int col = n0 + 16 * j + lr;
        C[(size_t)row * N + col] = acc[i][j][v];
      }
}

// ---------------- Kernel 3: RMSNorm(Q,K over H) + RoPE + head split ----------------

__global__ __launch_bounds__(256) void swa_norm_rope(const float* __restrict__ qkv,
                                                     const float* __restrict__ qw,
                                                     const float* __restrict__ kw,
                                                     bf16* __restrict__ qh,
                                                     bf16* __restrict__ kh,
                                                     bf16* __restrict__ vT) {
  int m = blockIdx.x;          // b*S + s
  int b = m >> 11;
  int s = m & (S_LEN - 1);
  const float* row = qkv + (size_t)m * (3 * HID);

  float sq = 0.f, sk = 0.f;
  for (int j = threadIdx.x; j < HID; j += 256) {
    float x = row[j];        sq += x * x;
    float y = row[HID + j];  sk += y * y;
  }
  __shared__ float red[2][256];
  red[0][threadIdx.x] = sq;
  red[1][threadIdx.x] = sk;
  __syncthreads();
  for (int off = 128; off > 0; off >>= 1) {
    if (threadIdx.x < off) {
      red[0][threadIdx.x] += red[0][threadIdx.x + off];
      red[1][threadIdx.x] += red[1][threadIdx.x + off];
    }
    __syncthreads();
  }
  float rq = rsqrtf(red[0][0] * (1.f / HID) + 1e-6f);
  float rk = rsqrtf(red[1][0] * (1.f / HID) + 1e-6f);
  const float qscale = 0.08838834764831845f;  // 1/sqrt(128)

  for (int p = threadIdx.x; p < NHEAD * (HDIM / 2); p += 256) {
    int h = p >> 6, d = p & 63;
    float inv_freq = __expf((float)d * (-9.210340371976184f / 64.f));  // 10000^(-d/64)
    float fr = (float)s * inv_freq, sn, cs;
    __sincosf(fr, &sn, &cs);
    int i1 = h * HDIM + d, i2 = i1 + 64;
    float q1 = row[i1] * rq * qw[i1];
    float q2 = row[i2] * rq * qw[i2];
    float k1 = row[HID + i1] * rk * kw[i1];
    float k2 = row[HID + i2] * rk * kw[i2];
    size_t o = ((size_t)(b * NHEAD + h) * S_LEN + s) * HDIM;
    qh[o + d]      = (bf16)((q1 * cs - q2 * sn) * qscale);
    qh[o + d + 64] = (bf16)((q2 * cs + q1 * sn) * qscale);
    kh[o + d]      = (bf16)(k1 * cs - k2 * sn);
    kh[o + d + 64] = (bf16)(k2 * cs + k1 * sn);
  }
  for (int j = threadIdx.x; j < HID; j += 256) {
    int h = j >> 7, d = j & 127;
    vT[((size_t)(b * NHEAD + h) * HDIM + d) * S_LEN + s] = (bf16)row[2 * HID + j];
  }
}

// ---------------- Kernel 4: sliding-window flash attention ----------------
// grid (S/64, NH, B), block 128 (4 waves). Loop bounds block-uniform => barriers legal.

__global__ __launch_bounds__(128) void swa_attn(const bf16* __restrict__ Q,
                                                const bf16* __restrict__ Kh,
                                                const bf16* __restrict__ vT,
                                                bf16* __restrict__ Ao) {
  int b = blockIdx.z, h = blockIdx.y;
  int q_blk = blockIdx.x * 64;
  int w = threadIdx.x >> 5;
  int lane = threadIdx.x & 31, lr = lane & 15, half = lane >> 4;
  int q0 = q_blk + w * 16;

  const bf16* qh = Q  + (size_t)(b * NHEAD + h) * S_LEN * HDIM;
  const bf16* kh = Kh + (size_t)(b * NHEAD + h) * S_LEN * HDIM;
  const bf16* vh = vT + (size_t)(b * NHEAD + h) * HDIM * S_LEN;

  bf16x16 aq[4];
#pragma unroll
  for (int t = 0; t < 4; t++) aq[t] = load_a_frag(qh, HDIM, q0, 32 * t);

  f32x8 acc[8] = {};
  float rowM[8], rowL[8];
#pragma unroll
  for (int v = 0; v < 8; v++) { rowM[v] = -1e30f; rowL[v] = 0.f; }

  __shared__ bf16 plds[4][16 * 32];

  int k_lo = q_blk - (WIN - 1);
  if (k_lo < 0) k_lo = 0;
  k_lo &= ~31;
  int k_hi = q_blk + 63;

  for (int kb = k_lo; kb <= k_hi; kb += 32) {
    f32x8 s0 = {}, s1 = {};
#pragma unroll
    for (int t = 0; t < 4; t++) {
      bf16x16 b0 = load_b_frag(kh, HDIM, kb, 32 * t);
      bf16x16 b1 = load_b_frag(kh, HDIM, kb + 16, 32 * t);
      s0 = wmma_bf16(aq[t], b0, s0);
      s1 = wmma_bf16(aq[t], b1, s1);
    }
#pragma unroll
    for (int v = 0; v < 8; v++) {
      int qi = q0 + v + 8 * half;
      int k0i = kb + lr, k1i = kb + 16 + lr;
      float x0 = ((qi >= k0i) && (qi - k0i < WIN)) ? s0[v] : -1e30f;
      float x1 = ((qi >= k1i) && (qi - k1i < WIN)) ? s1[v] : -1e30f;
      float mx = fmaxf(x0, x1);
#pragma unroll
      for (int d = 1; d < 16; d <<= 1) mx = fmaxf(mx, __shfl_xor(mx, d, 32));
      float newM = fmaxf(rowM[v], mx);
      float scale = __expf(rowM[v] - newM);
      rowM[v] = newM;
      float p0 = __expf(x0 - newM);
      float p1 = __expf(x1 - newM);
      float ps = p0 + p1;
#pragma unroll
      for (int d = 1; d < 16; d <<= 1) ps += __shfl_xor(ps, d, 32);
      rowL[v] = rowL[v] * scale + ps;
#pragma unroll
      for (int t = 0; t < 8; t++) acc[t][v] *= scale;
      int r = v + 8 * half;
      plds[w][r * 32 + lr]      = (bf16)p0;
      plds[w][r * 32 + 16 + lr] = (bf16)p1;
    }
    __syncthreads();
    bf16x16 ap = load_a_frag(&plds[w][0], 32, 0, 0);
    __syncthreads();
#pragma unroll
    for (int t = 0; t < 8; t++) {
      bf16x16 bv = load_b_frag(vh, S_LEN, 16 * t, kb);
      acc[t] = wmma_bf16(ap, bv, acc[t]);
    }
  }

#pragma unroll
  for (int v = 0; v < 8; v++) {
    float inv = 1.f / rowL[v];
    int s_idx = q0 + v + 8 * half;
#pragma unroll
    for (int t = 0; t < 8; t++) {
      float val = acc[t][v] * inv;
      Ao[((size_t)b * S_LEN + s_idx) * HID + h * HDIM + 16 * t + lr] = (bf16)val;
    }
  }
}

// ---------------- launch ----------------

extern "C" void kernel_launch(void* const* d_in, const int* in_sizes, int n_in,
                              void* d_out, int out_size, void* d_ws, size_t ws_size,
                              hipStream_t stream) {
  const float* hs   = (const float*)d_in[0];  // [2,2048,2048]
  const float* wqkv = (const float*)d_in[1];  // [6144,2048]
  const float* qnw  = (const float*)d_in[2];  // [2048]
  const float* knw  = (const float*)d_in[3];  // [2048]
  const float* wo   = (const float*)d_in[4];  // [2048,2048]
  float* out = (float*)d_out;                 // [2,2048,2048]

  char* ws = (char*)d_ws;
  const size_t N_HS   = (size_t)2 * S_LEN * HID;       // 8,388,608
  const size_t N_WQKV = (size_t)3 * HID * HID;         // 12,582,912
  const size_t N_WO   = (size_t)HID * HID;             // 4,194,304

  bf16*  hs_bf   = (bf16*)(ws + 0);                          // 16 MB (reused as attn out)
  bf16*  wqkv_bf = (bf16*)(ws + 16777216);                   // 24 MB
  bf16*  wo_bf   = (bf16*)(ws + 41943040);                   // 8 MB
  float* qkv_f   = (float*)(ws + 50331648);                  // 96 MB
  bf16*  q_bf    = (bf16*)(ws + 150994944);                  // 16 MB
  bf16*  k_bf    = (bf16*)(ws + 167772160);                  // 16 MB
  bf16*  vT_bf   = (bf16*)(ws + 184549376);                  // 16 MB
  bf16*  attn_bf = hs_bf;  // alias: hs no longer needed after QKV GEMM

  swa_cvt_bf16<<<(int)((N_HS + 255) / 256), 256, 0, stream>>>(hs, hs_bf, (int)N_HS);
  swa_cvt_bf16<<<(int)((N_WQKV + 255) / 256), 256, 0, stream>>>(wqkv, wqkv_bf, (int)N_WQKV);
  swa_cvt_bf16<<<(int)((N_WO + 255) / 256), 256, 0, stream>>>(wo, wo_bf, (int)N_WO);

  // QKV GEMM: [4096,6144] = [4096,2048] x [6144,2048]^T
  swa_gemm_bf16<<<dim3(6144 / 128, 4096 / 128), 256, 0, stream>>>(
      hs_bf, wqkv_bf, qkv_f, 4096, 6144, 2048);

  swa_norm_rope<<<4096, 256, 0, stream>>>(qkv_f, qnw, knw, q_bf, k_bf, vT_bf);

  swa_attn<<<dim3(S_LEN / 64, NHEAD, 2), 128, 0, stream>>>(q_bf, k_bf, vT_bf, attn_bf);

  // output projection: [4096,2048] = [4096,2048] x [2048,2048]^T
  swa_gemm_bf16<<<dim3(2048 / 128, 4096 / 128), 256, 0, stream>>>(
      attn_bf, wo_bf, out, 4096, 2048, 2048);
}
